// LSTMModel_11742440587681
// MI455X (gfx1250) — matrix operand order repaired and checked by reading, not compiled, not run on hardware
//
#include <hip/hip_runtime.h>

// ---------------------------------------------------------------------------
// LSTM on MI455X (gfx1250): persistent 4-WG kernel, W in LDS, bf16 WMMA.
//   D=128, H=256, B=64, T=2048, G=4H=1024
// Each WG owns a 64-column slice of H and computes all 4 gates for it:
//   W_hh slice [256 x 256] bf16 (128 KB LDS) + W_ih slice [128 x 256] (64 KB)
// Per step: x_t tile arrives via GLOBAL_LOAD_ASYNC_TO_LDS (double buffered,
// issued one step ahead), h broadcast returns via the Tensor Data Mover
// (TENSOR_LOAD_TO_LDS with a 1-D D#), gates via v_wmma_f32_16x16x32_bf16,
// c stays in VGPRs in WMMA C-fragment layout. 4 blocks sync with a monotonic
// atomic grid barrier.
// ---------------------------------------------------------------------------

typedef __attribute__((ext_vector_type(16))) __bf16 bf16x16;
typedef __attribute__((ext_vector_type(8)))  __bf16 bf16x8;
typedef __attribute__((ext_vector_type(8)))  float  f32x8;
typedef __attribute__((ext_vector_type(4)))  unsigned u32x4;
typedef __attribute__((ext_vector_type(8)))  int      i32x8;
typedef __attribute__((ext_vector_type(4)))  int      i32x4;

typedef __attribute__((address_space(1))) i32x4 g_i32x4;   // global int4
typedef __attribute__((address_space(3))) i32x4 l_i32x4;   // LDS int4

#define LSTM_D   128
#define LSTM_H   256
#define LSTM_B   64
#define LSTM_T   2048
#define LSTM_G   1024      // 4*H
#define NWG      4
#define TPB      256       // 8 waves of 32

// LDS element counts
#define WHH_ELEMS (16 * 8 * 512)   // 16 n-tiles * 8 k-tiles * (32 lanes * 16)
#define WIH_ELEMS (16 * 4 * 512)
#define HBUF_ELEMS (LSTM_B * LSTM_H)
#define XBUF_ELEMS (LSTM_B * LSTM_D)
#define SMEM_BYTES ((WHH_ELEMS + WIH_ELEMS + HBUF_ELEMS + 2 * XBUF_ELEMS) * 2)

#if defined(__has_builtin)
#if __has_builtin(__builtin_amdgcn_global_load_async_to_lds_b128)
#define HAS_ASYNC_LDS 1
#endif
#if __has_builtin(__builtin_amdgcn_tensor_load_to_lds) && \
    __has_builtin(__builtin_amdgcn_s_wait_tensorcnt)
#define HAS_TDM 1
#endif
#endif

__device__ __forceinline__ float sigmoidf_(float x) {
    return 1.0f / (1.0f + __expf(-x));
}

__device__ __forceinline__ void wait_asynccnt0() {
#if defined(__has_builtin) && __has_builtin(__builtin_amdgcn_s_wait_asynccnt)
    __builtin_amdgcn_s_wait_asynccnt(0);
#else
    asm volatile("s_wait_asynccnt 0" ::: "memory");
#endif
}

// ---------------------------------------------------------------------------
// Pack W_ih / W_hh (fp32 row-major [K, 4H]) into per-WG, per-tile, per-lane
// bf16 WMMA B-fragment order:
//   pack[wg][nt][kt][lane][e]  = W[kt*32 + (lane>>4)*16 + e][gcol]
//   gcol = q*H + wg*64 + jl, with ns = nt*16 + (lane&15), q = ns>>6, jl = ns&63
// so a wave loads one B fragment as a single 32-byte read at lane*32.
// ---------------------------------------------------------------------------
__global__ void lstm_pack_weights(const float* __restrict__ W_ih,
                                  const float* __restrict__ W_hh,
                                  __bf16* __restrict__ whh_pack,
                                  __bf16* __restrict__ wih_pack) {
    int i = blockIdx.x * blockDim.x + threadIdx.x;
    const int WHH_N = NWG * WHH_ELEMS;   // 262144
    const int WIH_N = NWG * WIH_ELEMS;   // 131072
    if (i < WHH_N) {
        int e    = i & 15;
        int lane = (i >> 4) & 31;
        int kt   = (i >> 9) & 7;
        int nt   = (i >> 12) & 15;
        int wg   = i >> 16;
        int k    = kt * 32 + ((lane >> 4) << 4) + e;
        int ns   = nt * 16 + (lane & 15);
        int q    = ns >> 6;
        int jl   = ns & 63;
        int gcol = q * LSTM_H + wg * 64 + jl;
        whh_pack[i] = (__bf16)W_hh[k * LSTM_G + gcol];
    } else if (i < WHH_N + WIH_N) {
        int j    = i - WHH_N;
        int e    = j & 15;
        int lane = (j >> 4) & 31;
        int kt   = (j >> 9) & 3;
        int nt   = (j >> 11) & 15;
        int wg   = j >> 15;
        int k    = kt * 32 + ((lane >> 4) << 4) + e;
        int ns   = nt * 16 + (lane & 15);
        int q    = ns >> 6;
        int jl   = ns & 63;
        int gcol = q * LSTM_H + wg * 64 + jl;
        wih_pack[j] = (__bf16)W_ih[k * LSTM_G + gcol];
    }
}

// Whole-sequence x: fp32 -> bf16 (keeps layout [B, T, D]); one-time, HBM-bound.
__global__ void lstm_convert_x(const float* __restrict__ x,
                               __bf16* __restrict__ xb, int n4) {
    int i = blockIdx.x * blockDim.x + threadIdx.x;   // float4 index
    if (i < n4) {
        float4 v = ((const float4*)x)[i];
        __bf16* d = xb + (size_t)i * 4;
        d[0] = (__bf16)v.x; d[1] = (__bf16)v.y;
        d[2] = (__bf16)v.z; d[3] = (__bf16)v.w;
    }
}

__global__ void lstm_init_ws(unsigned* bar) {
    if (threadIdx.x == 0 && blockIdx.x == 0) *bar = 0u;
}

// ---------------------------------------------------------------------------
// Stage the bf16 x tile for timestep t into an LDS buffer (async if possible).
// 1024 chunks of 16 B; thread handles chunks tid, tid+256, ...
// ---------------------------------------------------------------------------
__device__ __forceinline__ void stage_x_tile(const __bf16* __restrict__ xb,
                                             __bf16* __restrict__ dst, int t,
                                             int tid) {
    for (int c = tid; c < LSTM_B * LSTM_D / 8; c += TPB) {
        int b = c >> 4;                 // 16 chunks per batch row
        int o = (c & 15) * 8;
        const __bf16* gp = xb + (size_t)b * (LSTM_T * LSTM_D)
                              + (size_t)t * LSTM_D + o;
        __bf16* lp = dst + b * LSTM_D + o;
#if defined(HAS_ASYNC_LDS)
        __builtin_amdgcn_global_load_async_to_lds_b128(
            (g_i32x4*)gp, (l_i32x4*)lp, 0, 0);
#else
        *(uint4*)lp = *(const uint4*)gp;
#endif
    }
}

// ---------------------------------------------------------------------------
// Persistent recurrent kernel.
// ---------------------------------------------------------------------------
__global__ void __launch_bounds__(TPB, 1)
lstm_persistent(const __bf16* __restrict__ xb,     // [B, T, D] bf16
                const float* __restrict__ bias,    // [4H]
                const float* __restrict__ Wout,    // [H, 1]
                const float* __restrict__ bout,    // [1]
                const __bf16* __restrict__ whh_pack,
                const __bf16* __restrict__ wih_pack,
                __bf16* __restrict__ ws_h,         // [2][B][H] bf16 double buffer
                unsigned* __restrict__ bar,
                float* __restrict__ out) {         // [B]
    extern __shared__ __align__(32) char smem[];
    __bf16* s_whh = (__bf16*)smem;                  // 65536 elems (128 KB)
    __bf16* s_wih = s_whh + WHH_ELEMS;              // 32768 elems ( 64 KB)
    __bf16* s_h   = s_wih + WIH_ELEMS;              // 16384 elems ( 32 KB)
    __bf16* s_x   = s_h + HBUF_ELEMS;               // 2*8192 elems ( 32 KB)

    const int wg   = blockIdx.x;
    const int tid  = threadIdx.x;
    const int lane = tid & 31;
    const int wave = tid >> 5;
    const int mt   = wave >> 1;                     // M tile (batch rows /16)
    const int j0   = wg * 64;                       // this WG's H-column base

    // ---- stage weights into LDS, zero h(t=0), preload x(0) ---------------
    stage_x_tile(xb, s_x, 0, tid);
    {
        const uint4* srcA = (const uint4*)(whh_pack + (size_t)wg * WHH_ELEMS);
        uint4* dstA = (uint4*)s_whh;
        for (int i = tid; i < WHH_ELEMS * 2 / 16; i += TPB) dstA[i] = srcA[i];
        const uint4* srcB = (const uint4*)(wih_pack + (size_t)wg * WIH_ELEMS);
        uint4* dstB = (uint4*)s_wih;
        for (int i = tid; i < WIH_ELEMS * 2 / 16; i += TPB) dstB[i] = srcB[i];
        uint4 z; z.x = z.y = z.z = z.w = 0u;
        uint4* hz = (uint4*)s_h;
        for (int i = tid; i < HBUF_ELEMS * 2 / 16; i += TPB) hz[i] = z;
    }

    // ---- per-combo state: this wave owns combos (mt, jn0) and (mt, jn1) --
    int jn_[2];
    jn_[0] = (wave * 2) & 3;
    jn_[1] = (wave * 2 + 1) & 3;
    f32x8 cstate[2];
    float biasr[2][4];
    #pragma unroll
    for (int s = 0; s < 2; ++s) {
        #pragma unroll
        for (int r = 0; r < 8; ++r) cstate[s][r] = 0.0f;
        #pragma unroll
        for (int q = 0; q < 4; ++q) {
            int gcol = q * LSTM_H + j0 + jn_[s] * 16 + (lane & 15);
            biasr[s][q] = bias[gcol];
        }
    }
#if defined(HAS_ASYNC_LDS)
    wait_asynccnt0();
#endif
    __syncthreads();

    const bf16x16* whhv = (const bf16x16*)s_whh;    // [nt][kt][32 lane-chunks]
    const bf16x16* wihv = (const bf16x16*)s_wih;

    const int row  = mt * 16 + (lane & 15);         // batch row for A frags
    const int klo  = (lane >> 4) * 16;              // A-fragment K sub-offset

    for (int t = 0; t < LSTM_T; ++t) {
        const __bf16* sxc = s_x + (t & 1) * XBUF_ELEMS;

        // ---- kick off async copy of x(t+1) into the other buffer ---------
        if (t + 1 < LSTM_T)
            stage_x_tile(xb, s_x + ((t + 1) & 1) * XBUF_ELEMS, t + 1, tid);

        // ---- A fragments (shared by both combos: same mt) ----------------
        bf16x16 ah[8], ax[4];
        #pragma unroll
        for (int kt = 0; kt < 8; ++kt) {
            const __bf16* p = s_h + row * LSTM_H + kt * 32 + klo;
            bf16x8 lo = *(const bf16x8*)p;
            bf16x8 hi = *(const bf16x8*)(p + 8);
            ah[kt] = __builtin_shufflevector(lo, hi, 0,1,2,3,4,5,6,7,
                                                      8,9,10,11,12,13,14,15);
        }
        #pragma unroll
        for (int kt = 0; kt < 4; ++kt) {
            const __bf16* p = sxc + row * LSTM_D + kt * 32 + klo;
            bf16x8 lo = *(const bf16x8*)p;
            bf16x8 hi = *(const bf16x8*)(p + 8);
            ax[kt] = __builtin_shufflevector(lo, hi, 0,1,2,3,4,5,6,7,
                                                      8,9,10,11,12,13,14,15);
        }

        __bf16* hdst = ws_h + (size_t)(t & 1) * (LSTM_B * LSTM_H);

        // ---- 2 combos x 4 gates x 12 K-tiles of WMMA ---------------------
        #pragma unroll
        for (int s = 0; s < 2; ++s) {
            const int jn = jn_[s];
            f32x8 acc[4];
            #pragma unroll
            for (int q = 0; q < 4; ++q) {
                const int nt = q * 4 + jn;
                f32x8 a;
                #pragma unroll
                for (int r = 0; r < 8; ++r) a[r] = biasr[s][q];
                #pragma unroll
                for (int kt = 0; kt < 4; ++kt) {
                    bf16x16 bf = wihv[(nt * 4 + kt) * 32 + lane];
                    a = __builtin_amdgcn_wmma_f32_16x16x32_bf16(
                            false, ax[kt], false, bf, (short)0, a, false, false);
                }
                #pragma unroll
                for (int kt = 0; kt < 8; ++kt) {
                    bf16x16 bf = whhv[(nt * 8 + kt) * 32 + lane];
                    a = __builtin_amdgcn_wmma_f32_16x16x32_bf16(
                            false, ah[kt], false, bf, (short)0, a, false, false);
                }
                acc[q] = a;
            }
            // ---- in-register LSTM cell update (C-fragment layout) --------
            const int colg = j0 + jn * 16 + (lane & 15);
            f32x8 c = cstate[s];
            #pragma unroll
            for (int r = 0; r < 8; ++r) {
                float iv = sigmoidf_(acc[0][r]);
                float fv = sigmoidf_(acc[1][r]);
                float gv = tanhf(acc[2][r]);
                float ov = sigmoidf_(acc[3][r]);
                float cv = fv * c[r] + iv * gv;
                c[r] = cv;
                float hv = ov * tanhf(cv);
                int brow = mt * 16 + r + ((lane >> 4) << 3);
                hdst[brow * LSTM_H + colg] = (__bf16)hv;
            }
            cstate[s] = c;
        }

        // ---- grid barrier across the 4 WGs (monotonic counter) -----------
        __threadfence();
        __syncthreads();
        if (tid == 0) {
            atomicAdd(bar, 1u);
            const unsigned target = (unsigned)(t + 1) * NWG;
            volatile unsigned* vb = bar;
            while (*vb < target) __builtin_amdgcn_s_sleep(1);
        }
        __syncthreads();
        __threadfence();

        // ---- pull the full new h [64 x 256] into LDS ---------------------
#if defined(HAS_TDM)
        if (tid < 32) {   // one wave drives the Tensor Data Mover
            const __bf16* hsrc = ws_h + (size_t)(t & 1) * (LSTM_B * LSTM_H);
            unsigned lds_addr = (unsigned)(unsigned long long)(void*)s_h;
            unsigned long long ga = (unsigned long long)(const void*)hsrc;
            // D# group 0: count=1, lds_addr, global_addr[56:0], type=2
            u32x4 g0;
            g0[0] = 1u;
            g0[1] = lds_addr;
            g0[2] = (unsigned)(ga & 0xffffffffu);
            g0[3] = (unsigned)((ga >> 32) & 0x01ffffffu) | 0x80000000u;
            // D# group 1: 1-D copy of 16384 bf16 elements (data_size=2B)
            i32x8 g1;
            g1[0] = (int)(1u << 16);          // data_size=1 (2 bytes)
            g1[1] = (int)(16384u << 16);      // tensor_dim0[15:0]  -> [31:16]
            g1[2] = (int)(1u << 16);          // dim0 hi=0; tensor_dim1=1
            g1[3] = (int)(16384u << 16);      // dim1 hi=0; tile_dim0=16384
            g1[4] = 1;                        // tile_dim1=1, tile_dim2=0
            g1[5] = 16384;                    // tensor_dim0_stride[31:0]
            g1[6] = 0;
            g1[7] = 0;
            i32x4 g2; g2[0] = 0; g2[1] = 0; g2[2] = 0; g2[3] = 0;
            i32x4 g3; g3[0] = 0; g3[1] = 0; g3[2] = 0; g3[3] = 0;
            i32x8 g4; g4[0] = 0; g4[1] = 0; g4[2] = 0; g4[3] = 0;
                      g4[4] = 0; g4[5] = 0; g4[6] = 0; g4[7] = 0;
            __builtin_amdgcn_tensor_load_to_lds(g0, g1, g2, g3, g4, 0);
            __builtin_amdgcn_s_wait_tensorcnt(0);
        }
#else
        {
            const uint4* src = (const uint4*)(ws_h +
                               (size_t)(t & 1) * (LSTM_B * LSTM_H));
            uint4* dst = (uint4*)s_h;
            for (int i = tid; i < HBUF_ELEMS * 2 / 16; i += TPB) dst[i] = src[i];
        }
#endif
#if defined(HAS_ASYNC_LDS)
        wait_asynccnt0();   // x(t+1) tile now resident
#endif
        __syncthreads();
    }

    // ---- final projection: out[b] = h_T[b,:] . Wout + bout ---------------
    if (wg == 0 && tid < LSTM_B) {
        float acc = bout[0];
        #pragma unroll 4
        for (int j = 0; j < LSTM_H; ++j)
            acc += (float)s_h[tid * LSTM_H + j] * Wout[j];
        out[tid] = acc;
    }
}

// ---------------------------------------------------------------------------
extern "C" void kernel_launch(void* const* d_in, const int* in_sizes, int n_in,
                              void* d_out, int out_size, void* d_ws, size_t ws_size,
                              hipStream_t stream) {
    (void)in_sizes; (void)n_in; (void)out_size; (void)ws_size;
    const float* x    = (const float*)d_in[0];   // [B, T, D]
    const float* W_ih = (const float*)d_in[1];   // [D, 4H]
    const float* W_hh = (const float*)d_in[2];   // [H, 4H]
    const float* b    = (const float*)d_in[3];   // [4H]
    const float* Wout = (const float*)d_in[4];   // [H, 1]
    const float* bout = (const float*)d_in[5];   // [1]
    float* out = (float*)d_out;                  // [B, 1] -> 64 floats

    char* ws = (char*)d_ws;
    unsigned* bar   = (unsigned*)ws;                               // 256 B
    __bf16* ws_h    = (__bf16*)(ws + 256);                         // 64 KB
    __bf16* whh_pk  = (__bf16*)(ws + 256 + 65536);                 // 512 KB
    __bf16* wih_pk  = (__bf16*)(ws + 256 + 65536 + 524288);        // 256 KB
    __bf16* xb      = (__bf16*)(ws + 256 + 65536 + 524288 + 262144); // 32 MB

    lstm_init_ws<<<1, 64, 0, stream>>>(bar);

    const int np = NWG * WHH_ELEMS + NWG * WIH_ELEMS;              // 393216
    lstm_pack_weights<<<(np + 255) / 256, 256, 0, stream>>>(
        W_ih, W_hh, whh_pk, wih_pk);

    const int n4 = LSTM_B * LSTM_T * LSTM_D / 4;                   // 4194304
    lstm_convert_x<<<(n4 + 255) / 256, 256, 0, stream>>>(x, xb, n4);

    lstm_persistent<<<NWG, TPB, SMEM_BYTES, stream>>>(
        xb, b, Wout, bout, whh_pk, wih_pk, ws_h, bar, out);
}
